// MoEMLP_15247133900938
// MI455X (gfx1250) — compile-verified, matrix operand order checked by reading, not run on hardware
//
#include <hip/hip_runtime.h>
#include <hip/hip_bf16.h>

#define T_TOK 4096
#define H_DIM 1024
#define FF_DIM 4096
#define NE 8

typedef __attribute__((ext_vector_type(16))) __bf16 v16bf;
typedef __attribute__((ext_vector_type(8)))  float  v8f;

union FragBF {
    v16bf v;
    uint4 q[2];
};

__device__ __forceinline__ unsigned short f2bf(float f) {
    unsigned int u = __float_as_uint(f);
    u += 0x7FFFu + ((u >> 16) & 1u);   // round-to-nearest-even
    return (unsigned short)(u >> 16);
}

// gelu(x) = x * sigmoid(2 * c * (x + 0.044715 x^3)) -- branch-free tanh approx
__device__ __forceinline__ float gelu_tanh(float x) {
    const float c0 = 0.7978845608028654f;  // sqrt(2/pi)
    float u = c0 * (x + 0.044715f * x * x * x);
    return x / (1.0f + __expf(-2.0f * u));
}

// ---------------------------------------------------------------- utilities
__global__ void moe_zero_stats(float* stats) {
    if (threadIdx.x < 17) stats[threadIdx.x] = 0.0f;
}

__global__ __launch_bounds__(256)
void moe_cvt_x(const float* __restrict__ x, unsigned short* __restrict__ xbf) {
    int i = blockIdx.x * 256 + threadIdx.x;          // float4 index
    float4 v = ((const float4*)x)[i];
    uint2 o;
    o.x = (unsigned int)f2bf(v.x) | ((unsigned int)f2bf(v.y) << 16);
    o.y = (unsigned int)f2bf(v.z) | ((unsigned int)f2bf(v.w) << 16);
    *(uint2*)&xbf[(size_t)i * 4] = o;
}

// ---------------------------------------------------------------- router
__global__ __launch_bounds__(256)
void moe_router(const float* __restrict__ x, const float* __restrict__ Wg,
                float* __restrict__ logits_out, float* __restrict__ combine,
                float* __restrict__ stats) {
    const int lane = threadIdx.x & 31;
    const int wave = threadIdx.x >> 5;
    const int t = blockIdx.x * 8 + wave;           // one wave32 per token

    float p[NE];
#pragma unroll
    for (int e = 0; e < NE; ++e) p[e] = 0.0f;
    for (int h = lane; h < H_DIM; h += 32) {
        float xv = x[(size_t)t * H_DIM + h];
        const float* wr = &Wg[h * NE];
#pragma unroll
        for (int e = 0; e < NE; ++e) p[e] += xv * wr[e];
    }
#pragma unroll
    for (int off = 16; off > 0; off >>= 1)
#pragma unroll
        for (int e = 0; e < NE; ++e) p[e] += __shfl_down(p[e], off, 32);

    if (lane == 0) {
        float l[NE];
#pragma unroll
        for (int e = 0; e < NE; ++e) { l[e] = p[e]; logits_out[t * NE + e] = p[e]; }

        // top-2 (first-occurrence tie-break, matching lax.top_k)
        int i0 = 0; float v0 = l[0];
        for (int e = 1; e < NE; ++e) if (l[e] > v0) { v0 = l[e]; i0 = e; }
        int i1 = 0; float v1 = -3.4e38f;
        for (int e = 0; e < NE; ++e) if (e != i0 && l[e] > v1) { v1 = l[e]; i1 = e; }

        float e1 = __expf(v1 - v0);
        float inv = 1.0f / (1.0f + e1);
        float w0 = inv, w1 = e1 * inv;

        float crow[NE];
#pragma unroll
        for (int e = 0; e < NE; ++e) crow[e] = 0.0f;
        crow[i0] = w0; crow[i1] = w1;
#pragma unroll
        for (int e = 0; e < NE; ++e) combine[t * NE + e] = crow[e];

        // full softmax stats + z-loss
        float m = v0, s = 0.0f;
#pragma unroll
        for (int e = 0; e < NE; ++e) s += __expf(l[e] - m);
        float rs = 1.0f / s;
#pragma unroll
        for (int e = 0; e < NE; ++e) atomicAdd(&stats[e], __expf(l[e] - m) * rs);
        float lse = m + __logf(s);
        atomicAdd(&stats[16], lse * lse);
        atomicAdd(&stats[8 + i0], 1.0f);
        atomicAdd(&stats[8 + i1], 1.0f);
    }
}

// ---------------------------------------------------------------- shared GEMM mainloop
// Block: 128 threads = 4 wave32. Block tile 128(M) x 64(N), K-step 32.
// Each wave owns 32(M) x 64(N): 2 A frags, 4 B frags, 8 WMMAs per K-step.
// A (bf16, row-major in ws) staged with GLOBAL_LOAD_ASYNC_TO_LDS_B128 (ASYNCcnt).
// B (fp32 weights) converted to bf16 + transposed into LDS on the VALU path.
template <int KDIM, int NSTRIDE>
__device__ __forceinline__ void gemm_mainloop(const unsigned short* __restrict__ Abf,
                                              const float* __restrict__ Bf,
                                              int m0, int n0, v8f acc[2][4]) {
    __shared__ __align__(16) unsigned short As[128][32];  // [m][k] bf16
    __shared__ __align__(16) unsigned short Bs[64][32];   // B^T: [n][k] bf16
    const int tid  = threadIdx.x;
    const int lane = tid & 31;
    const int wave = tid >> 5;
    const int half = lane >> 4;
    const int mrow = lane & 15;
    const int kOff = half * 8;

    for (int k0 = 0; k0 < KDIM; k0 += 32) {
        // A: 128x32 bf16 = 512 16B chunks -> async DMA straight into LDS
#pragma unroll
        for (int r = 0; r < 4; ++r) {
            int id = r * 128 + tid;
            int m = id >> 2, c = id & 3;
            unsigned lds = (unsigned)(size_t)&As[m][c * 8];   // LDS aperture: addr[31:0]
            unsigned long long g =
                (unsigned long long)&Abf[(size_t)(m0 + m) * KDIM + k0 + c * 8];
            asm volatile("global_load_async_to_lds_b128 %0, %1, off"
                         :: "v"(lds), "v"(g) : "memory");
        }
        // B: 32x64 fp32 -> bf16, transposed into Bs[n][k]
#pragma unroll
        for (int r = 0; r < 16; ++r) {
            int id = r * 128 + tid;
            int k = id >> 6, n = id & 63;
            Bs[n][k] = f2bf(Bf[(size_t)(k0 + k) * NSTRIDE + n0 + n]);
        }
        if (k0 + 32 < KDIM)   // pull next weight K-slab toward L2/L0
            __builtin_prefetch(&Bf[(size_t)(k0 + 32) * NSTRIDE + n0 + (tid & 63)], 0, 0);

        asm volatile("s_wait_asynccnt 0x0" ::: "memory");
        __syncthreads();

        FragBF a[2];
#pragma unroll
        for (int mi = 0; mi < 2; ++mi) {
            const unsigned short* ar = As[wave * 32 + mi * 16 + mrow];
            a[mi].q[0] = *(const uint4*)&ar[kOff];
            a[mi].q[1] = *(const uint4*)&ar[16 + kOff];
        }
#pragma unroll
        for (int j = 0; j < 4; ++j) {
            FragBF b;
            const unsigned short* br = Bs[j * 16 + mrow];
            b.q[0] = *(const uint4*)&br[kOff];
            b.q[1] = *(const uint4*)&br[16 + kOff];
#pragma unroll
            for (int mi = 0; mi < 2; ++mi)
                acc[mi][j] = __builtin_amdgcn_wmma_f32_16x16x32_bf16(
                    false, a[mi].v, false, b.v, (short)0, acc[mi][j], false, false);
        }
        __syncthreads();
    }
}

// ---------------------------------------------------------------- GEMM1: h = gelu(x @ W1[e])
__global__ __launch_bounds__(128)
void moe_gemm1(const unsigned short* __restrict__ xbf,   // [T, H] bf16
               const float* __restrict__ W1e,            // [H, FF] fp32
               unsigned short* __restrict__ hbf)         // [T, FF] bf16
{
    const int lane = threadIdx.x & 31;
    const int wave = threadIdx.x >> 5;
    const int m0 = blockIdx.y * 128;
    const int n0 = blockIdx.x * 64;
    const int half = lane >> 4;
    const int mrow = lane & 15;

    v8f acc[2][4] = {};
    gemm_mainloop<H_DIM, FF_DIM>(xbf, W1e, m0, n0, acc);

#pragma unroll
    for (int mi = 0; mi < 2; ++mi)
#pragma unroll
        for (int j = 0; j < 4; ++j) {
            int col = n0 + j * 16 + mrow;
#pragma unroll
            for (int v = 0; v < 8; ++v) {
                int row = m0 + wave * 32 + mi * 16 + v + half * 8;
                hbf[(size_t)row * FF_DIM + col] = f2bf(gelu_tanh(acc[mi][j][v]));
            }
        }
}

// ---------------------------------------------------------------- GEMM2: out (+)= combine[:,e] * (h @ W2[e])
__global__ __launch_bounds__(128)
void moe_gemm2(const unsigned short* __restrict__ hbf,   // [T, FF] bf16
               const float* __restrict__ W2e,            // [FF, H] fp32
               const float* __restrict__ combine,        // [T, NE]
               float* __restrict__ out,                  // [T, H]
               int expert, int first)
{
    const int lane = threadIdx.x & 31;
    const int wave = threadIdx.x >> 5;
    const int m0 = blockIdx.y * 128;
    const int n0 = blockIdx.x * 64;
    const int half = lane >> 4;
    const int mrow = lane & 15;

    v8f acc[2][4] = {};
    gemm_mainloop<FF_DIM, H_DIM>(hbf, W2e, m0, n0, acc);

#pragma unroll
    for (int mi = 0; mi < 2; ++mi)
#pragma unroll
        for (int j = 0; j < 4; ++j) {
            int col = n0 + j * 16 + mrow;
#pragma unroll
            for (int v = 0; v < 8; ++v) {
                int row = m0 + wave * 32 + mi * 16 + v + half * 8;
                float cw = combine[row * NE + expert];
                float val = cw * acc[mi][j][v];
                size_t idx = (size_t)row * H_DIM + col;
                if (first) out[idx] = val;
                else       out[idx] += val;
            }
        }
}

// ---------------------------------------------------------------- aux loss
__global__ void moe_finalize(const float* __restrict__ stats, float* __restrict__ aux_out) {
    float sumA = 0.0f, sumF = 0.0f;
    for (int e = 0; e < NE; ++e) { sumA += stats[e]; sumF += stats[8 + e]; }
    float sw = 0.0f;
    for (int e = 0; e < NE; ++e) sw += (stats[e] / sumA) * (stats[8 + e] / sumF);
    float switch_loss = (float)NE * sw;
    float z_loss = stats[16] / (float)T_TOK;
    *aux_out = switch_loss + 0.1f * z_loss;
}

// ---------------------------------------------------------------- launcher
extern "C" void kernel_launch(void* const* d_in, const int* in_sizes, int n_in,
                              void* d_out, int out_size, void* d_ws, size_t ws_size,
                              hipStream_t stream) {
    const float* x  = (const float*)d_in[0];   // [T, H]
    const float* Wg = (const float*)d_in[1];   // [H, E]
    const float* W1 = (const float*)d_in[2];   // [E, H, FF]
    const float* W2 = (const float*)d_in[3];   // [E, FF, H]

    float* out    = (float*)d_out;                          // [T, H]
    float* logits = out + (size_t)T_TOK * H_DIM;            // [T, E]
    float* aux    = logits + (size_t)T_TOK * NE;            // scalar

    char* ws = (char*)d_ws;
    unsigned short* xbf = (unsigned short*)ws;                                   // 8 MB
    unsigned short* hbf = (unsigned short*)(ws + (size_t)T_TOK * H_DIM * 2);     // 33.5 MB
    float* combine = (float*)(ws + (size_t)T_TOK * H_DIM * 2 +
                                   (size_t)T_TOK * FF_DIM * 2);                  // 128 KB
    float* stats = combine + (size_t)T_TOK * NE;                                 // 17 floats

    moe_zero_stats<<<1, 32, 0, stream>>>(stats);
    moe_cvt_x<<<(T_TOK * H_DIM / 4) / 256, 256, 0, stream>>>(x, xbf);
    moe_router<<<T_TOK / 8, 256, 0, stream>>>(x, Wg, logits, combine, stats);

    for (int e = 0; e < NE; ++e) {
        moe_gemm1<<<dim3(FF_DIM / 64, T_TOK / 128), 128, 0, stream>>>(
            xbf, W1 + (size_t)e * H_DIM * FF_DIM, hbf);
        moe_gemm2<<<dim3(H_DIM / 64, T_TOK / 128), 128, 0, stream>>>(
            hbf, W2 + (size_t)e * FF_DIM * H_DIM, combine, out, e, e == 0);
    }
    moe_finalize<<<1, 1, 0, stream>>>(stats, aux);
}
